// MultiheadAttention_18262200943004
// MI455X (gfx1250) — matrix operand order repaired and checked by reading, not compiled, not run on hardware
//
#include <hip/hip_runtime.h>
#include <hip/hip_bf16.h>

typedef __attribute__((ext_vector_type(16))) _Float16 v16h;
typedef __attribute__((ext_vector_type(8)))  _Float16 v8h;
typedef __attribute__((ext_vector_type(8)))  float    v8f;

#define WMMA_F16(a, b, c) \
  __builtin_amdgcn_wmma_f32_16x16x32_f16(false, (a), false, (b), (short)0, (c), false, false)

static __device__ __forceinline__ v16h combine16(v8h lo, v8h hi) {
  return __builtin_shufflevector(lo, hi, 0, 1, 2, 3, 4, 5, 6, 7, 8, 9, 10, 11, 12, 13, 14, 15);
}
static __device__ __forceinline__ v8h ldsv8(const _Float16* p) {
  return *(const v8h*)p;   // 16B-aligned by construction -> ds_load_b128
}

// ---------------- constants ----------------
// B=2, S=2048, E=1024, H=16, hd=64, rows = B*S = 4096
#define ROWS   4096
#define EDIM   1024
#define QKVN   3072
#define SEQ    2048
#define NHEAD  16
#define HD     64

// ---------------- f32 -> f16 convert ----------------
__global__ void cvt_f32_f16(const float* __restrict__ in, _Float16* __restrict__ out, int n) {
  int i = (blockIdx.x * blockDim.x + threadIdx.x) * 4;
  if (i + 3 < n) {
    out[i + 0] = (_Float16)in[i + 0];
    out[i + 1] = (_Float16)in[i + 1];
    out[i + 2] = (_Float16)in[i + 2];
    out[i + 3] = (_Float16)in[i + 3];
  }
}

// ---------------- QKV GEMM: [4096,1024]f16 x [1024,3072]f16 + bias -> Q/K/V head-major f16 ----
__global__ __launch_bounds__(256) void qkv_gemm(const _Float16* __restrict__ X,
                                                const _Float16* __restrict__ W,
                                                const float* __restrict__ bias,
                                                _Float16* __restrict__ Qb,
                                                _Float16* __restrict__ Kb,
                                                _Float16* __restrict__ Vb) {
  const int ASTR = 40, BSTR = 40;            // strides in halves; *2B -> 16B multiples
  __shared__ __align__(16) _Float16 Asm[128 * 40];   // A tile row-major [m][k]
  __shared__ __align__(16) _Float16 BsmT[64 * 40];   // B tile transposed [n][k]

  const int t = threadIdx.x;
  const int lane = t & 31, wave = t >> 5;
  const int half = lane >> 4, l15 = lane & 15;
  const int wm = wave >> 1, wn = wave & 1;   // 4 x 2 wave grid
  const int gm0 = blockIdx.x * 128;
  const int gn0 = blockIdx.y * 64;

  v8f acc[2][2];
#pragma unroll
  for (int i = 0; i < 2; i++)
#pragma unroll
    for (int j = 0; j < 2; j++)
#pragma unroll
      for (int e = 0; e < 8; e++) acc[i][j][e] = 0.0f;

  for (int k0 = 0; k0 < EDIM; k0 += 32) {
    __syncthreads();
    {   // A tile: 128x32, 16 halves/thread, coalesced; contiguous LDS stores
      int r = t >> 1, c = (t & 1) * 16;
      const _Float16* src = X + (long)(gm0 + r) * EDIM + k0 + c;
      _Float16* dst = Asm + r * ASTR + c;
      if (k0 + 32 < EDIM) __builtin_prefetch(src + 32, 0, 3);
#pragma unroll
      for (int i = 0; i < 16; i++) dst[i] = src[i];
    }
    {   // B tile: 32x64 from W (coalesced), stored transposed [n][k]
      int r = t >> 3, c = (t & 7) * 8;
      const _Float16* src = W + (long)(k0 + r) * QKVN + gn0 + c;
      if (k0 + 32 < EDIM) __builtin_prefetch(src + 32 * QKVN, 0, 3);
#pragma unroll
      for (int i = 0; i < 8; i++) BsmT[(c + i) * BSTR + r] = src[i];
    }
    __syncthreads();

    v16h af[2], bf[2];
#pragma unroll
    for (int mt = 0; mt < 2; mt++) {
      const _Float16* ap = Asm + (wm * 32 + mt * 16 + l15) * ASTR + half * 8;
      af[mt] = combine16(ldsv8(ap), ldsv8(ap + 16));
    }
#pragma unroll
    for (int nt = 0; nt < 2; nt++) {
      const _Float16* bp = BsmT + (wn * 32 + nt * 16 + l15) * BSTR + half * 8;
      bf[nt] = combine16(ldsv8(bp), ldsv8(bp + 16));
    }
#pragma unroll
    for (int mt = 0; mt < 2; mt++)
#pragma unroll
      for (int nt = 0; nt < 2; nt++)
        acc[mt][nt] = WMMA_F16(af[mt], bf[nt], acc[mt][nt]);
  }

  // epilogue: bias add + scatter into head-major Q/K/V (fuses reshape/transpose/split)
#pragma unroll
  for (int mt = 0; mt < 2; mt++)
#pragma unroll
    for (int nt = 0; nt < 2; nt++)
#pragma unroll
      for (int r = 0; r < 8; r++) {
        int gr = gm0 + wm * 32 + mt * 16 + r + 8 * half;
        int gc = gn0 + wn * 32 + nt * 16 + l15;
        float v = acc[mt][nt][r] + bias[gc];
        int head = gc / 192;
        int within = gc - head * 192;
        int type = within >> 6;          // 0=Q 1=K 2=V
        int d = within & 63;
        int bi = gr >> 11;               // / SEQ
        int s = gr & 2047;
        long idx = (((long)(bi * NHEAD + head)) * SEQ + s) * HD + d;
        _Float16 hv = (_Float16)v;
        if (type == 0)      Qb[idx] = hv;
        else if (type == 1) Kb[idx] = hv;
        else                Vb[idx] = hv;
      }
}

// ---------------- flash attention: per block 64 q-rows of one (b,h) ----------------
__global__ __launch_bounds__(128) void flash_attn(const _Float16* __restrict__ Qb,
                                                  const _Float16* __restrict__ Kb,
                                                  const _Float16* __restrict__ Vb,
                                                  _Float16* __restrict__ Ctx) {
  const int STR = 72;                                  // 144B rows: 16B multiple
  __shared__ __align__(16) _Float16 Qs[64 * 72];       // [q][d]
  __shared__ __align__(16) _Float16 Ks[64 * 72];       // [key][d]  (== K^T for B-frags)
  __shared__ __align__(16) _Float16 VsT[64 * 72];      // [d][key]  (B-frags contiguous)
  __shared__ __align__(16) _Float16 Ps[4 * 16 * 72];   // per-wave P patch [q][key]

  const int t = threadIdx.x;
  const int lane = t & 31, wave = t >> 5;
  const int half = lane >> 4, l15 = lane & 15;
  const int qt0 = blockIdx.x * 64;
  const int bh = blockIdx.y;                       // b*16 + h
  const long base = (long)bh * SEQ * HD;

  {   // Q tile 64x64, 32 halves/thread
    int r = t >> 1, c = (t & 1) * 32;
    const _Float16* src = Qb + base + (long)(qt0 + r) * HD + c;
    _Float16* dst = Qs + r * STR + c;
#pragma unroll
    for (int i = 0; i < 32; i++) dst[i] = src[i];
  }

  float m_i[8], l_i[8];
  v8f o[4];
#pragma unroll
  for (int r = 0; r < 8; r++) { m_i[r] = -1e30f; l_i[r] = 0.0f; }
#pragma unroll
  for (int dt = 0; dt < 4; dt++)
#pragma unroll
    for (int e = 0; e < 8; e++) o[dt][e] = 0.0f;

  __syncthreads();
  v16h qa[2];                                      // Q A-frags, rows = wave*16 + l15
  {
    const _Float16* qp = Qs + (wave * 16 + l15) * STR;
#pragma unroll
    for (int kc = 0; kc < 2; kc++) {
      const _Float16* p = qp + kc * 32 + half * 8;
      qa[kc] = combine16(ldsv8(p), ldsv8(p + 16));
    }
  }

  const float scale = 0.125f;                      // 1/sqrt(64)

  for (int kt = 0; kt < SEQ; kt += 64) {
    __syncthreads();
    {   // K row-major, V transposed into LDS
      int r = t >> 1, c = (t & 1) * 32;
      const _Float16* ks = Kb + base + (long)(kt + r) * HD + c;
      const _Float16* vs = Vb + base + (long)(kt + r) * HD + c;
      _Float16* kd = Ks + r * STR + c;
#pragma unroll
      for (int i = 0; i < 32; i++) kd[i] = ks[i];
#pragma unroll
      for (int i = 0; i < 32; i++) VsT[(c + i) * STR + r] = vs[i];
    }
    __syncthreads();

    // S = Q·K^T : 4 key sub-tiles x 2 d-chunks  (all frag reads = ds_load_b128 pairs)
    v8f sf[4];
#pragma unroll
    for (int nt = 0; nt < 4; nt++) {
#pragma unroll
      for (int e = 0; e < 8; e++) sf[nt][e] = 0.0f;
#pragma unroll
      for (int kc = 0; kc < 2; kc++) {
        const _Float16* kp = Ks + (nt * 16 + l15) * STR + kc * 32 + half * 8;
        v16h bfrag = combine16(ldsv8(kp), ldsv8(kp + 16));
        sf[nt] = WMMA_F16(qa[kc], bfrag, sf[nt]);
      }
    }
#pragma unroll
    for (int nt = 0; nt < 4; nt++)
#pragma unroll
      for (int e = 0; e < 8; e++) sf[nt][e] *= scale;

    // online softmax: row stats per (half-group, r); reduce across 16 lanes
#pragma unroll
    for (int r = 0; r < 8; r++) {
      float mx = fmaxf(fmaxf(sf[0][r], sf[1][r]), fmaxf(sf[2][r], sf[3][r]));
#pragma unroll
      for (int off = 8; off >= 1; off >>= 1) mx = fmaxf(mx, __shfl_xor(mx, off, 32));
      float mnew = fmaxf(m_i[r], mx);
      float corr = __expf(m_i[r] - mnew);
      m_i[r] = mnew;
      float rs = 0.0f;
#pragma unroll
      for (int nt = 0; nt < 4; nt++) {
        float p = __expf(sf[nt][r] - mnew);
        sf[nt][r] = p;
        rs += p;
      }
#pragma unroll
      for (int off = 8; off >= 1; off >>= 1) rs += __shfl_xor(rs, off, 32);
      l_i[r] = l_i[r] * corr + rs;
#pragma unroll
      for (int dt = 0; dt < 4; dt++) o[dt][r] *= corr;
    }

    // C-frag -> A-frag re-layout through per-wave LDS patch (stores strided, loads contiguous)
    _Float16* pw = Ps + wave * 16 * STR;
#pragma unroll
    for (int nt = 0; nt < 4; nt++)
#pragma unroll
      for (int r = 0; r < 8; r++)
        pw[(r + 8 * half) * STR + nt * 16 + l15] = (_Float16)sf[nt][r];
    __syncthreads();

    // O += P·V
#pragma unroll
    for (int kc = 0; kc < 2; kc++) {
      const _Float16* pp = Ps + (wave * 16 + l15) * STR + kc * 32 + half * 8;
      v16h pa = combine16(ldsv8(pp), ldsv8(pp + 16));
#pragma unroll
      for (int dt = 0; dt < 4; dt++) {
        const _Float16* vp = VsT + (dt * 16 + l15) * STR + kc * 32 + half * 8;
        v16h vfrag = combine16(ldsv8(vp), ldsv8(vp + 16));
        o[dt] = WMMA_F16(pa, vfrag, o[dt]);
      }
    }
  }

  // finalize: divide by l, store context in [B*S, E] layout (fuses transpose back)
  const int bb = bh >> 4, h = bh & 15;
#pragma unroll
  for (int r = 0; r < 8; r++) {
    float inv = 1.0f / l_i[r];
    int qrow = qt0 + wave * 16 + r + 8 * half;
    long orow = ((long)bb * SEQ + qrow) * EDIM + h * HD;
#pragma unroll
    for (int dt = 0; dt < 4; dt++)
      Ctx[orow + dt * 16 + l15] = (_Float16)(o[dt][r] * inv);
  }
}

// ---------------- output GEMM: [4096,1024]f16 x [1024,1024]f16 + bias -> f32 ----------------
__global__ __launch_bounds__(256) void out_gemm(const _Float16* __restrict__ A,
                                                const _Float16* __restrict__ W,
                                                const float* __restrict__ bias,
                                                float* __restrict__ out) {
  const int ASTR = 40, BSTR = 40;
  __shared__ __align__(16) _Float16 Asm[128 * 40];
  __shared__ __align__(16) _Float16 BsmT[64 * 40];

  const int t = threadIdx.x;
  const int lane = t & 31, wave = t >> 5;
  const int half = lane >> 4, l15 = lane & 15;
  const int wm = wave >> 1, wn = wave & 1;
  const int gm0 = blockIdx.x * 128;
  const int gn0 = blockIdx.y * 64;

  v8f acc[2][2];
#pragma unroll
  for (int i = 0; i < 2; i++)
#pragma unroll
    for (int j = 0; j < 2; j++)
#pragma unroll
      for (int e = 0; e < 8; e++) acc[i][j][e] = 0.0f;

  for (int k0 = 0; k0 < EDIM; k0 += 32) {
    __syncthreads();
    {
      int r = t >> 1, c = (t & 1) * 16;
      const _Float16* src = A + (long)(gm0 + r) * EDIM + k0 + c;
      _Float16* dst = Asm + r * ASTR + c;
      if (k0 + 32 < EDIM) __builtin_prefetch(src + 32, 0, 3);
#pragma unroll
      for (int i = 0; i < 16; i++) dst[i] = src[i];
    }
    {
      int r = t >> 3, c = (t & 7) * 8;
      const _Float16* src = W + (long)(k0 + r) * EDIM + gn0 + c;
      if (k0 + 32 < EDIM) __builtin_prefetch(src + 32 * EDIM, 0, 3);
#pragma unroll
      for (int i = 0; i < 8; i++) BsmT[(c + i) * BSTR + r] = src[i];
    }
    __syncthreads();

    v16h af[2], bf[2];
#pragma unroll
    for (int mt = 0; mt < 2; mt++) {
      const _Float16* ap = Asm + (wm * 32 + mt * 16 + l15) * ASTR + half * 8;
      af[mt] = combine16(ldsv8(ap), ldsv8(ap + 16));
    }
#pragma unroll
    for (int nt = 0; nt < 2; nt++) {
      const _Float16* bp = BsmT + (wn * 32 + nt * 16 + l15) * BSTR + half * 8;
      bf[nt] = combine16(ldsv8(bp), ldsv8(bp + 16));
    }
#pragma unroll
    for (int mt = 0; mt < 2; mt++)
#pragma unroll
      for (int nt = 0; nt < 2; nt++)
        acc[mt][nt] = WMMA_F16(af[mt], bf[nt], acc[mt][nt]);
  }

#pragma unroll
  for (int mt = 0; mt < 2; mt++)
#pragma unroll
    for (int nt = 0; nt < 2; nt++)
#pragma unroll
      for (int r = 0; r < 8; r++) {
        int gr = gm0 + wm * 32 + mt * 16 + r + 8 * half;
        int gc = gn0 + wn * 32 + nt * 16 + l15;
        out[(long)gr * EDIM + gc] = acc[mt][nt][r] + bias[gc];
      }
}

// ---------------- launch ----------------
extern "C" void kernel_launch(void* const* d_in, const int* in_sizes, int n_in,
                              void* d_out, int out_size, void* d_ws, size_t ws_size,
                              hipStream_t stream) {
  const float* x    = (const float*)d_in[0];   // [2,2048,1024]
  const float* Wqkv = (const float*)d_in[1];   // [1024,3072]
  const float* bqkv = (const float*)d_in[2];   // [3072]
  const float* Wo   = (const float*)d_in[3];   // [1024,1024]
  const float* bo   = (const float*)d_in[4];   // [1024]
  float* out = (float*)d_out;

  char* ws = (char*)d_ws;
  _Float16* Xh    = (_Float16*)(ws);                         // 8 MB
  _Float16* Wqkvh = (_Float16*)(ws + 8388608);               // 6 MB
  _Float16* Woh   = (_Float16*)(ws + 14680064);              // 2 MB
  _Float16* Qb    = (_Float16*)(ws + 16777216);              // 8 MB
  _Float16* Kb    = (_Float16*)(ws + 25165824);               // 8 MB
  _Float16* Vb    = (_Float16*)(ws + 33554432);              // 8 MB
  _Float16* Ctx   = (_Float16*)(ws + 41943040);              // 8 MB

  // convert inputs to f16
  cvt_f32_f16<<<4096, 256, 0, stream>>>(x,    Xh,    ROWS * EDIM);
  cvt_f32_f16<<<3072, 256, 0, stream>>>(Wqkv, Wqkvh, EDIM * QKVN);
  cvt_f32_f16<<<1024, 256, 0, stream>>>(Wo,   Woh,   EDIM * EDIM);

  // qkv projection + head split
  qkv_gemm<<<dim3(ROWS / 128, QKVN / 64), 256, 0, stream>>>(Xh, Wqkvh, bqkv, Qb, Kb, Vb);

  // flash attention: 32 q-tiles x 32 (b,h)
  flash_attn<<<dim3(SEQ / 64, 2 * NHEAD), 128, 0, stream>>>(Qb, Kb, Vb, Ctx);

  // output projection
  out_gemm<<<dim3(ROWS / 128, EDIM / 64), 256, 0, stream>>>(Ctx, Woh, bo, out);
}
